// STDALapShot_33406255629074
// MI455X (gfx1250) — matrix operand order repaired and checked by reading, not compiled, not run on hardware
//
#include <hip/hip_runtime.h>
#include <hip/hip_bf16.h>
#include <math.h>

// Problem constants (match reference setup_inputs)
#define NQ    4096      // query count
#define NS    1600      // support count
#define DIMF  1024      // feature dim
#define NC    64        // classes
#define KNN   12        // neighbors kept (log2(4096))
#define KSEL  13        // k+1 including self
#define EMAX  128       // symmetric edge-list capacity per row
#define NBINS 4096      // median histogram bins over [0,128)

// LDS-tiled Gram GEMM geometry
#define BM 128          // macro-tile M (4 waves)
#define BN 64           // macro-tile N (2 waves)
#define BK 32           // K-slab per step
#define APITCH 40       // LDS row pitch in bf16 elems (80 B, 16B-aligned, bank-spread)
#define BPITCH 40

typedef __attribute__((ext_vector_type(16))) __bf16 v16bf;
typedef __attribute__((ext_vector_type(8)))  float  v8f;

union Frag { v16bf v; uint4 q[2]; };

__device__ __forceinline__ unsigned short f2bf(float f) {
    unsigned u = __float_as_uint(f);
    u += 0x7FFFu + ((u >> 16) & 1u);   // round-to-nearest-even
    return (unsigned short)(u >> 16);
}

// ---------------- async global->LDS copy (CDNA5 ASYNCcnt path), with fallback
typedef int v4i_gcc __attribute__((__vector_size__(4 * sizeof(int))));
typedef __attribute__((address_space(1))) v4i_gcc* as1_v4i_p;   // global
typedef __attribute__((address_space(3))) v4i_gcc* as3_v4i_p;   // LDS

#if __has_builtin(__builtin_amdgcn_global_load_async_to_lds_b128)
#define HAVE_ASYNC_LDS 1
__device__ __forceinline__ void async_copy_b128(void* lds_dst, const void* glb_src) {
    __builtin_amdgcn_global_load_async_to_lds_b128(
        (as1_v4i_p)(glb_src),
        (as3_v4i_p)(lds_dst), 0, 0);
}
__device__ __forceinline__ void wait_async_lds() {
#if __has_builtin(__builtin_amdgcn_s_wait_asynccnt)
    __builtin_amdgcn_s_wait_asynccnt(0);
#else
    asm volatile("s_wait_asynccnt 0" ::: "memory");
#endif
}
#else
#define HAVE_ASYNC_LDS 0
__device__ __forceinline__ void async_copy_b128(void* lds_dst, const void* glb_src) {
    *(uint4*)lds_dst = *(const uint4*)glb_src;   // reg-staged: global_load_b128 + ds_store_b128
}
__device__ __forceinline__ void wait_async_lds() {}
#endif

// ---------------------------------------------------------------- prototypes
__global__ void k_protos(const float* __restrict__ fs, const int* __restrict__ ys,
                         float* __restrict__ protos) {
    int c = blockIdx.x;          // class
    int t = threadIdx.x;         // 256 threads, 4 cols each
    float a0 = 0.f, a1 = 0.f, a2 = 0.f, a3 = 0.f, cnt = 0.f;
    for (int r = 0; r < NS; ++r) {
        if (ys[r] == c) {
            const float* row = fs + (size_t)r * DIMF;
            a0 += row[t];       a1 += row[t + 256];
            a2 += row[t + 512]; a3 += row[t + 768];
            cnt += 1.f;
        }
    }
    float inv = 1.f / fmaxf(cnt, 1.f);
    float* p = protos + (size_t)c * DIMF;
    p[t] = a0 * inv; p[t + 256] = a1 * inv; p[t + 512] = a2 * inv; p[t + 768] = a3 * inv;
}

// ------------------------------------------------- f32 -> bf16 + row sq-norm
__global__ void k_bf16_norm(const float* __restrict__ X, unsigned short* __restrict__ Xb,
                            float* __restrict__ norms, int rows) {
    int r = blockIdx.x;
    if (r >= rows) return;
    int t = threadIdx.x;
    const float* row = X + (size_t)r * DIMF;
    unsigned short* rb = Xb + (size_t)r * DIMF;
    float ss = 0.f;
    for (int j = t; j < DIMF; j += 256) {
        float v = row[j];
        ss += v * v;
        rb[j] = f2bf(v);
    }
    __shared__ float red[256];
    red[t] = ss; __syncthreads();
    for (int s = 128; s > 0; s >>= 1) { if (t < s) red[t] += red[t + s]; __syncthreads(); }
    if (t == 0) norms[r] = red[0];
}

// -------------------------------------------- WMMA GEMM: out[M][N] = X * Y^T
// Direct-from-global variant (used for the small 4096x64 proto GEMM).
__global__ void k_gemm_xyT(const unsigned short* __restrict__ X,
                           const unsigned short* __restrict__ Y,
                           float* __restrict__ out, int M, int N, int ldo) {
    const int K = DIMF;
    int wid  = (blockIdx.x * blockDim.x + threadIdx.x) >> 5;
    int lane = threadIdx.x & 31;
    int tn_cnt = N >> 5;
    int tiles  = (M >> 5) * tn_cnt;
    if (wid >= tiles) return;
    int m0 = (wid / tn_cnt) << 5;
    int n0 = (wid % tn_cnt) << 5;
    int half = lane >> 4, lr = lane & 15;

    v8f acc00 = {}, acc01 = {}, acc10 = {}, acc11 = {};
    const unsigned short* xa0 = X + (size_t)(m0 + lr)      * K;
    const unsigned short* xa1 = X + (size_t)(m0 + 16 + lr) * K;
    const unsigned short* yb0 = Y + (size_t)(n0 + lr)      * K;
    const unsigned short* yb1 = Y + (size_t)(n0 + 16 + lr) * K;

    for (int kb = 0; kb < K; kb += 32) {
        if (kb + 32 < K) {
            __builtin_prefetch(xa0 + kb + 32, 0, 3);
            __builtin_prefetch(yb0 + kb + 32, 0, 3);
        }
        Frag a0, a1, b0, b1;
        a0.q[0] = *(const uint4*)(xa0 + kb + half * 8);
        a0.q[1] = *(const uint4*)(xa0 + kb + 16 + half * 8);
        a1.q[0] = *(const uint4*)(xa1 + kb + half * 8);
        a1.q[1] = *(const uint4*)(xa1 + kb + 16 + half * 8);
        b0.q[0] = *(const uint4*)(yb0 + kb + half * 8);
        b0.q[1] = *(const uint4*)(yb0 + kb + 16 + half * 8);
        b1.q[0] = *(const uint4*)(yb1 + kb + half * 8);
        b1.q[1] = *(const uint4*)(yb1 + kb + 16 + half * 8);

        acc00 = __builtin_amdgcn_wmma_f32_16x16x32_bf16(false, a0.v, false, b0.v, (short)0, acc00, false, false);
        acc01 = __builtin_amdgcn_wmma_f32_16x16x32_bf16(false, a0.v, false, b1.v, (short)0, acc01, false, false);
        acc10 = __builtin_amdgcn_wmma_f32_16x16x32_bf16(false, a1.v, false, b0.v, (short)0, acc10, false, false);
        acc11 = __builtin_amdgcn_wmma_f32_16x16x32_bf16(false, a1.v, false, b1.v, (short)0, acc11, false, false);
    }
    #pragma unroll
    for (int r = 0; r < 8; ++r) {
        int rm = r + half * 8;
        out[(size_t)(m0 + rm)      * ldo + n0 + lr]      = acc00[r];
        out[(size_t)(m0 + rm)      * ldo + n0 + 16 + lr] = acc01[r];
        out[(size_t)(m0 + 16 + rm) * ldo + n0 + lr]      = acc10[r];
        out[(size_t)(m0 + 16 + rm) * ldo + n0 + 16 + lr] = acc11[r];
    }
}

// --------------------- LDS-tiled, double-buffered Gram GEMM: out = X * X^T
// Block = 256 thr = 8 waves in a 4(M) x 2(N) grid; macro-tile 128x64; each
// wave computes a 32x32 tile (2x2 WMMA frags). Operand slabs staged in LDS
// via async global->LDS copies (ASYNCcnt) with double buffering.
__global__ void __launch_bounds__(256) k_gemm_lds(const unsigned short* __restrict__ X,
                                                  float* __restrict__ out) {
    __shared__ __align__(16) unsigned short sA[2][BM * APITCH];
    __shared__ __align__(16) unsigned short sB[2][BN * BPITCH];

    const int nb = NQ / BN;                   // 64 tiles along N
    int m0 = (blockIdx.x / nb) * BM;
    int n0 = (blockIdx.x % nb) * BN;
    int t = threadIdx.x;
    int wave = t >> 5, lane = t & 31;
    int wm = (wave & 3) << 5;                 // wave M offset within macro-tile
    int wn = (wave >> 2) << 5;                // wave N offset within macro-tile
    int half = lane >> 4, lr = lane & 15;

    // A slab: BM rows x BK cols bf16 = 64 B/row = 4 x b128 chunks -> 512 chunks
    // B slab: BN rows -> 256 chunks
    auto copyA = [&](int buf, int kb) {
        #pragma unroll
        for (int c = t; c < BM * 4; c += 256) {
            int row = c >> 2, sub = c & 3;
            async_copy_b128(&sA[buf][row * APITCH + sub * 8],
                            X + (size_t)(m0 + row) * DIMF + kb + sub * 8);
        }
    };
    auto copyB = [&](int buf, int kb) {
        int row = t >> 2, sub = t & 3;        // exactly 256 chunks
        async_copy_b128(&sB[buf][row * BPITCH + sub * 8],
                        X + (size_t)(n0 + row) * DIMF + kb + sub * 8);
    };

    v8f acc00 = {}, acc01 = {}, acc10 = {}, acc11 = {};

    copyA(0, 0);
    copyB(0, 0);
    const int KSTEPS = DIMF / BK;             // 32
    for (int s = 0; s < KSTEPS; ++s) {
        int b = s & 1;
        wait_async_lds();                     // own copies into buffer b landed
        __syncthreads();                      // everyone done reading buffer b^1
        if (s + 1 < KSTEPS) {                 // overlap next slab with compute
            copyA(b ^ 1, (s + 1) * BK);
            copyB(b ^ 1, (s + 1) * BK);
        }
        const unsigned short* arow0 = &sA[b][(wm + lr)      * APITCH];
        const unsigned short* arow1 = &sA[b][(wm + 16 + lr) * APITCH];
        const unsigned short* brow0 = &sB[b][(wn + lr)      * BPITCH];
        const unsigned short* brow1 = &sB[b][(wn + 16 + lr) * BPITCH];
        Frag a0, a1, b0f, b1f;
        a0.q[0]  = *(const uint4*)(arow0 + half * 8);
        a0.q[1]  = *(const uint4*)(arow0 + 16 + half * 8);
        a1.q[0]  = *(const uint4*)(arow1 + half * 8);
        a1.q[1]  = *(const uint4*)(arow1 + 16 + half * 8);
        b0f.q[0] = *(const uint4*)(brow0 + half * 8);
        b0f.q[1] = *(const uint4*)(brow0 + 16 + half * 8);
        b1f.q[0] = *(const uint4*)(brow1 + half * 8);
        b1f.q[1] = *(const uint4*)(brow1 + 16 + half * 8);

        acc00 = __builtin_amdgcn_wmma_f32_16x16x32_bf16(false, a0.v, false, b0f.v, (short)0, acc00, false, false);
        acc01 = __builtin_amdgcn_wmma_f32_16x16x32_bf16(false, a0.v, false, b1f.v, (short)0, acc01, false, false);
        acc10 = __builtin_amdgcn_wmma_f32_16x16x32_bf16(false, a1.v, false, b0f.v, (short)0, acc10, false, false);
        acc11 = __builtin_amdgcn_wmma_f32_16x16x32_bf16(false, a1.v, false, b1f.v, (short)0, acc11, false, false);
    }

    #pragma unroll
    for (int r = 0; r < 8; ++r) {
        int rm = r + half * 8;
        out[(size_t)(m0 + wm + rm)      * NQ + n0 + wn + lr]      = acc00[r];
        out[(size_t)(m0 + wm + rm)      * NQ + n0 + wn + 16 + lr] = acc01[r];
        out[(size_t)(m0 + wm + 16 + rm) * NQ + n0 + wn + lr]      = acc10[r];
        out[(size_t)(m0 + wm + 16 + rm) * NQ + n0 + wn + 16 + lr] = acc11[r];
    }
}

// ---------------------------- a = relu(|q|^2 + |p|^2 - 2 dot); d2min, d per row
__global__ void k_finalize_a(float* __restrict__ a, const float* __restrict__ qn,
                             const float* __restrict__ pn,
                             float* __restrict__ d2min, float* __restrict__ dval) {
    int i = blockIdx.x, c = threadIdx.x;   // 64 threads
    float dot = a[(size_t)i * NC + c];
    float v = fmaxf(qn[i] + pn[c] - 2.f * dot, 0.f);
    a[(size_t)i * NC + c] = v;
    __shared__ float red[64];
    red[c] = v; __syncthreads();
    for (int s = 32; s > 0; s >>= 1) { if (c < s) red[c] = fminf(red[c], red[c + s]); __syncthreads(); }
    if (c == 0) { d2min[i] = red[0]; dval[i] = sqrtf(red[0]); }
}

// ---------------------------------------------------------------- median(d)
__global__ void k_hist(const float* __restrict__ dval, int* __restrict__ hist) {
    int i = blockIdx.x * blockDim.x + threadIdx.x;
    if (i < NQ) {
        int b = (int)(dval[i] * 32.f);
        b = b < 0 ? 0 : (b > NBINS - 1 ? NBINS - 1 : b);
        atomicAdd(&hist[b], 1);
    }
}
__global__ void k_median(const int* __restrict__ hist, float* __restrict__ med) {
    int cum = 0;
    for (int b = 0; b < NBINS; ++b) {
        cum += hist[b];
        if (cum >= NQ / 2) { *med = ((float)b + 0.5f) / 32.f; return; }
    }
    *med = 64.f;
}
__global__ void k_lam(const float* __restrict__ d2, const float* __restrict__ med,
                      float* __restrict__ lam) {
    int i = blockIdx.x * blockDim.x + threadIdx.x;
    if (i < NQ) {
        float m = *med;
        lam[i] = __expf(-d2[i] / (2.f * m * m + 1e-8f));
    }
}

// ------------------------------------------ per-row top-13 over dist matrix
__global__ void k_topk(const float* __restrict__ dot, const float* __restrict__ qn,
                       int* __restrict__ nbrs, float* __restrict__ sigma) {
    int i = blockIdx.x, t = threadIdx.x;  // 256 threads per row
    unsigned long long best[KSEL];
    #pragma unroll
    for (int s = 0; s < KSEL; ++s) best[s] = 0xFFFFFFFFFFFFFFFFULL;
    float qni = qn[i];
    const float* row = dot + (size_t)i * NQ;
    for (int j = t; j < NQ; j += 256) {
        float sq = fmaxf(qni + qn[j] - 2.f * row[j], 0.f);
        float d  = sqrtf(sq);
        unsigned long long key = ((unsigned long long)__float_as_uint(d) << 32) | (unsigned)j;
        if (key < best[KSEL - 1]) {
            int p = KSEL - 1;
            while (p > 0 && best[p - 1] > key) { best[p] = best[p - 1]; --p; }
            best[p] = key;
        }
    }
    __shared__ unsigned long long sk[256 * KSEL];
    for (int s = 0; s < KSEL; ++s) sk[t * KSEL + s] = best[s];
    __syncthreads();
    for (int stride = 128; stride > 0; stride >>= 1) {
        if (t < stride) {
            unsigned long long* mine = &sk[t * KSEL];
            unsigned long long* oth  = &sk[(t + stride) * KSEL];
            for (int s = 0; s < KSEL; ++s) {
                unsigned long long key = oth[s];
                if (key < mine[KSEL - 1]) {
                    int p = KSEL - 1;
                    while (p > 0 && mine[p - 1] > key) { mine[p] = mine[p - 1]; --p; }
                    mine[p] = key;
                } else break;   // oth sorted ascending
            }
        }
        __syncthreads();
    }
    if (t == 0) {
        for (int s = 1; s <= KNN; ++s)
            nbrs[(size_t)i * KNN + (s - 1)] = (int)(sk[s] & 0xFFFFFFFFu);
        float d13 = __uint_as_float((unsigned)(sk[KSEL - 1] >> 32));
        sigma[i] = d13 + 1e-8f;
    }
}

// ----------------------- scatter symmetric half-weights into capped edge lists
__global__ void k_edges(const float* __restrict__ dot, const float* __restrict__ qn,
                        const int* __restrict__ nbrs, const float* __restrict__ sigma,
                        int* __restrict__ cnt, int* __restrict__ eidx, float* __restrict__ ew) {
    int g = blockIdx.x * blockDim.x + threadIdx.x;
    if (g >= NQ * KNN) return;
    int i = g / KNN, jj = g % KNN;
    int j = nbrs[(size_t)i * KNN + jj];
    float sq = fmaxf(qn[i] + qn[j] - 2.f * dot[(size_t)i * NQ + j], 0.f);
    float d  = sqrtf(sq);
    float h  = 0.5f * __expf(-d / (sigma[i] * sigma[j]));   // W = (W + W^T)/2
    int s1 = atomicAdd(&cnt[i], 1);
    if (s1 < EMAX) { eidx[(size_t)i * EMAX + s1] = j; ew[(size_t)i * EMAX + s1] = h; }
    int s2 = atomicAdd(&cnt[j], 1);
    if (s2 < EMAX) { eidx[(size_t)j * EMAX + s2] = i; ew[(size_t)j * EMAX + s2] = h; }
}

// ------------------------------------------------- coef = lam * D^-1 per row
__global__ void k_coef(const int* __restrict__ cnt, const float* __restrict__ ew,
                       const float* __restrict__ lam, float* __restrict__ coef) {
    int i = blockIdx.x * blockDim.x + threadIdx.x;
    if (i < NQ) {
        int n = cnt[i] < EMAX ? cnt[i] : EMAX;
        float s = 0.f;
        for (int e = 0; e < n; ++e) s += ew[(size_t)i * EMAX + e];
        coef[i] = lam[i] / (s + 1e-8f);
    }
}

// --------------------------------------------------------- y0 = softmax(-a)
__global__ void k_inity(const float* __restrict__ a, float* __restrict__ y) {
    int i = blockIdx.x, c = threadIdx.x;   // 64 threads
    float z = -a[(size_t)i * NC + c];
    __shared__ float red[64];
    red[c] = z; __syncthreads();
    for (int s = 32; s > 0; s >>= 1) { if (c < s) red[c] = fmaxf(red[c], red[c + s]); __syncthreads(); }
    float m = red[0]; __syncthreads();
    float e = __expf(z - m);
    red[c] = e; __syncthreads();
    for (int s = 32; s > 0; s >>= 1) { if (c < s) red[c] += red[c + s]; __syncthreads(); }
    y[(size_t)i * NC + c] = e / red[0];
}

// -------------------------- one MM iteration: y' = softmax(-a + coef * (W y))
__global__ void k_step(const float* __restrict__ a, const float* __restrict__ coef,
                       const int* __restrict__ cnt, const int* __restrict__ eidx,
                       const float* __restrict__ ew,
                       const float* __restrict__ yin, float* __restrict__ yout) {
    int i = blockIdx.x, c = threadIdx.x;   // 64 threads
    __shared__ int   se[EMAX];
    __shared__ float sw[EMAX];
    __shared__ float red[64];
    int n = cnt[i] < EMAX ? cnt[i] : EMAX;
    for (int e = c; e < n; e += 64) {
        se[e] = eidx[(size_t)i * EMAX + e];
        sw[e] = ew[(size_t)i * EMAX + e];
    }
    __syncthreads();
    float wy = 0.f;
    for (int e = 0; e < n; ++e) wy += sw[e] * yin[(size_t)se[e] * NC + c];
    float z = -a[(size_t)i * NC + c] + coef[i] * wy;
    red[c] = z; __syncthreads();
    for (int s = 32; s > 0; s >>= 1) { if (c < s) red[c] = fmaxf(red[c], red[c + s]); __syncthreads(); }
    float m = red[0]; __syncthreads();
    float e = __expf(z - m);
    red[c] = e; __syncthreads();
    for (int s = 32; s > 0; s >>= 1) { if (c < s) red[c] += red[c + s]; __syncthreads(); }
    yout[(size_t)i * NC + c] = e / red[0];
}

// ---------------------------------------------------------------- argmax
__global__ void k_argmax(const float* __restrict__ y, float* __restrict__ out) {
    int i = blockIdx.x * blockDim.x + threadIdx.x;
    if (i < NQ) {
        const float* r = y + (size_t)i * NC;
        int b = 0; float bv = r[0];
        for (int c = 1; c < NC; ++c) if (r[c] > bv) { bv = r[c]; b = c; }
        out[i] = (float)b;
    }
}

// =========================================================================
extern "C" void kernel_launch(void* const* d_in, const int* in_sizes, int n_in,
                              void* d_out, int out_size, void* d_ws, size_t ws_size,
                              hipStream_t stream) {
    const float* feat_s = (const float*)d_in[0];
    const int*   y_s    = (const int*)d_in[1];
    const float* feat_q = (const float*)d_in[2];

    char* ws = (char*)d_ws;
    size_t off = 0;
    auto alloc = [&](size_t bytes) -> void* {
        void* p = ws + off;
        off += (bytes + 255) & ~(size_t)255;
        return p;
    };

    float*          protos = (float*)         alloc((size_t)NC * DIMF * 4);
    unsigned short* qb     = (unsigned short*)alloc((size_t)NQ * DIMF * 2);
    unsigned short* pb     = (unsigned short*)alloc((size_t)NC * DIMF * 2);
    float*          qn     = (float*)         alloc((size_t)NQ * 4);
    float*          pn     = (float*)         alloc((size_t)NC * 4);
    float*          dotm   = (float*)         alloc((size_t)NQ * NQ * 4);   // 64 MB
    float*          abuf   = (float*)         alloc((size_t)NQ * NC * 4);
    float*          d2     = (float*)         alloc((size_t)NQ * 4);
    float*          dv     = (float*)         alloc((size_t)NQ * 4);
    float*          lam    = (float*)         alloc((size_t)NQ * 4);
    float*          sigma  = (float*)         alloc((size_t)NQ * 4);
    float*          coef   = (float*)         alloc((size_t)NQ * 4);
    int*            hist   = (int*)           alloc((size_t)NBINS * 4);
    float*          med    = (float*)         alloc(256);
    int*            nbrs   = (int*)           alloc((size_t)NQ * KNN * 4);
    int*            ecnt   = (int*)           alloc((size_t)NQ * 4);
    int*            eidx   = (int*)           alloc((size_t)NQ * EMAX * 4);
    float*          ewgt   = (float*)         alloc((size_t)NQ * EMAX * 4);
    float*          y0     = (float*)         alloc((size_t)NQ * NC * 4);
    float*          y1     = (float*)         alloc((size_t)NQ * NC * 4);

    // 1) prototypes and bf16 + norms
    k_protos   <<<NC, 256, 0, stream>>>(feat_s, y_s, protos);
    k_bf16_norm<<<NQ, 256, 0, stream>>>(feat_q, qb, qn, NQ);
    k_bf16_norm<<<NC, 256, 0, stream>>>(protos, pb, pn, NC);

    // 2) a = q . p^T  (4096x64, K=1024)
    {
        int tiles = (NQ >> 5) * (NC >> 5);
        int blocks = (tiles * 32 + 255) / 256;
        k_gemm_xyT<<<blocks, 256, 0, stream>>>(qb, pb, abuf, NQ, NC, NC);
    }
    k_finalize_a<<<NQ, 64, 0, stream>>>(abuf, qn, pn, d2, dv);

    // 3) lambda via histogram median of d
    (void)hipMemsetAsync(hist, 0, (size_t)NBINS * 4, stream);
    k_hist  <<<(NQ + 255) / 256, 256, 0, stream>>>(dv, hist);
    k_median<<<1, 1, 0, stream>>>(hist, med);
    k_lam   <<<(NQ + 255) / 256, 256, 0, stream>>>(d2, med, lam);

    // 4) Gram matrix q . q^T (4096x4096, K=1024), LDS-tiled + async copies
    {
        int blocks = (NQ / BM) * (NQ / BN);   // 32 * 64 = 2048
        k_gemm_lds<<<blocks, 256, 0, stream>>>(qb, dotm);
    }

    // 5) top-13 per row -> neighbors + sigma
    k_topk<<<NQ, 256, 0, stream>>>(dotm, qn, nbrs, sigma);

    // 6) symmetric sparse W and coef = lam * D^-1
    (void)hipMemsetAsync(ecnt, 0, (size_t)NQ * 4, stream);
    k_edges<<<(NQ * KNN + 255) / 256, 256, 0, stream>>>(dotm, qn, nbrs, sigma, ecnt, eidx, ewgt);
    k_coef <<<(NQ + 255) / 256, 256, 0, stream>>>(ecnt, ewgt, lam, coef);

    // 7) MM iterations (fixed 50 = MAX_ITER upper bound, deterministic)
    k_inity<<<NQ, 64, 0, stream>>>(abuf, y0);
    float* yin = y0; float* yout = y1;
    for (int it = 0; it < 50; ++it) {
        k_step<<<NQ, 64, 0, stream>>>(abuf, coef, ecnt, eidx, ewgt, yin, yout);
        float* tmp = yin; yin = yout; yout = tmp;
    }

    // 8) argmax -> output (class index as float)
    k_argmax<<<(NQ + 255) / 256, 256, 0, stream>>>(yin, (float*)d_out);
}
